// Block_46265387712766
// MI455X (gfx1250) — matrix operand order repaired and checked by reading, not compiled, not run on hardware
//
#include <hip/hip_runtime.h>

// ---------------------------------------------------------------------------
// Transformer block (pre-LN causal MHA + FFN) for gfx1250 (MI455X).
// All matmuls via v_wmma_f32_16x16x32_f16 (f16 A/B, f32 accumulate).
// Weights repacked into WMMA-fragment-major layout [K/32][N][32] so every
// B-fragment load is 2x global_load_b128. 64x64 wave tiles (16 WMMA / k-step).
// Attention is flash-style: online softmax, no SxS materialization.
// ---------------------------------------------------------------------------

typedef __attribute__((ext_vector_type(16))) _Float16 v16h;
typedef __attribute__((ext_vector_type(8)))  float    v8f;

#define WMMA_F16(a, b, c) \
  __builtin_amdgcn_wmma_f32_16x16x32_f16(false, (a), false, (b), (short)0, (c), false, false)

namespace cfg {
constexpr int BATCH = 64;
constexpr int SEQ   = 512;
constexpr int EMBED = 384;
constexpr int HEADS = 7;
constexpr int HS    = 54;               // true head size
constexpr int HSP   = 64;               // padded head size (mult. of 32)
constexpr int NO    = HEADS * HSP;      // 448  (padded concat width)
constexpr int NQKV  = 3 * NO;           // 1344 (Q|K|V packed columns)
constexpr int FFN   = 6 * EMBED;        // 2304
constexpr int M     = BATCH * SEQ;      // 32768 rows (tokens)
constexpr float QSCALE = 0.13608276348795434f; // 1/sqrt(54)
}

// ---------------------------------------------------------------------------
// WMMA fragment loaders (wave32; contiguous k order k = 16*(l/16) + e, same in A and B)
// ---------------------------------------------------------------------------

// A fragment: 16x32 f16. lane l: m = l&15; lanes 0-15 hold K in {0..15},
// lanes 16-31 hold K in {16..31}. Elements are 16 contiguous halves -> 2x b128.
__device__ __forceinline__ v16h load_afrag(const _Float16* __restrict__ p0, int rstride) {
  const unsigned l = threadIdx.x & 31u;
  const int m  = l & 15;
  const int kh = (l >> 4) << 4;                 // 0 or 16
  const _Float16* p = p0 + (size_t)m * rstride + kh;
  v16h r;
#pragma unroll
  for (int e = 0; e < 16; ++e)
    r[e] = p[e];                                // K=kh+e
  return r;
}

// B fragment from generic strides (used by attention on row-major Q/K/V).
// lane l: n = l&15; element e -> K-row = e + 16*(l>=16).
__device__ __forceinline__ v16h load_bfrag(const _Float16* __restrict__ p0,
                                           int kstride, int nstride) {
  const unsigned l = threadIdx.x & 31u;
  const int n  = l & 15;
  const int kh = (l >> 4) << 4;                 // 0 or 16
  const _Float16* p = p0 + (size_t)n * nstride + (size_t)kh * kstride;
  v16h r;
#pragma unroll
  for (int e = 0; e < 16; ++e)
    r[e] = p[(size_t)e * kstride];
  return r;
}

// B fragment from swizzled weights [K/32][N][32]: p0 -> (kt, n0, 0).
// Each lane reads 16 contiguous halves (32B) -> 2x global_load_b128.
__device__ __forceinline__ v16h load_bfrag_sw(const _Float16* __restrict__ p0) {
  const unsigned l = threadIdx.x & 31u;
  const int n  = l & 15;
  const int kh = (l >> 4) << 4;                 // 0 or 16
  const _Float16* p = p0 + n * 32 + kh;
  v16h r;
#pragma unroll
  for (int e = 0; e < 16; ++e)
    r[e] = p[e];
  return r;
}

// ---------------------------------------------------------------------------
// WMMA GEMM: C[M,N] = epilogue(A[M,KC] * Bsw), Bsw in [KC/32][N][32] layout.
// Block = 256 threads = 8 waves arranged 4(M) x 2(N); wave tile 64x64
// (16 accumulators -> 16 WMMAs per 4A+4B fragment loads). KC compile-time.
// ---------------------------------------------------------------------------
enum { EPI_BIAS_F16 = 0, EPI_BIAS_GELU_F16 = 1, EPI_BIAS_RES_F32 = 2 };

template <int EPI, int KC>
__global__ __launch_bounds__(256) void gemm_wmma_kernel(
    const _Float16* __restrict__ A, const _Float16* __restrict__ Bsw,
    const float* __restrict__ bias, const float* __restrict__ resid,
    void* __restrict__ Cout, int M, int N)
{
  const int wave = threadIdx.x >> 5;
  const int wm = wave >> 1;                     // 0..3
  const int wn = wave & 1;                      // 0..1
  const int m0 = blockIdx.y * 256 + wm * 64;
  const int n0 = blockIdx.x * 128 + wn * 64;
  if (m0 + 64 > M || n0 + 64 > N) return;       // no block-wide sync below

  v8f acc[4][4];
#pragma unroll
  for (int i = 0; i < 4; ++i)
#pragma unroll
    for (int j = 0; j < 4; ++j)
#pragma unroll
      for (int r = 0; r < 8; ++r) acc[i][j][r] = 0.0f;

#pragma unroll 2
  for (int k = 0; k < KC; k += 32) {
    if (k + 32 < KC) {  // speculative prefetch of next k-tile
      __builtin_prefetch(A + (size_t)m0 * KC + k + 32, 0, 0);
      __builtin_prefetch(Bsw + ((size_t)((k + 32) >> 5) * N + n0) * 32, 0, 0);
    }
    v16h a[4], b[4];
#pragma unroll
    for (int i = 0; i < 4; ++i)
      a[i] = load_afrag(A + (size_t)(m0 + 16 * i) * KC + k, KC);
    const _Float16* Bt = Bsw + ((size_t)(k >> 5) * N + n0) * 32;
#pragma unroll
    for (int j = 0; j < 4; ++j)
      b[j] = load_bfrag_sw(Bt + j * 16 * 32);
#pragma unroll
    for (int i = 0; i < 4; ++i)
#pragma unroll
      for (int j = 0; j < 4; ++j)
        acc[i][j] = WMMA_F16(a[i], b[j], acc[i][j]);
  }

  // Epilogue. C fragment layout: lane l -> n = l&15, row = r + 8*(l>>4).
  const unsigned l = threadIdx.x & 31u;
  const int cn  = l & 15;
  const int chi = (l >> 4) << 3;
#pragma unroll
  for (int ti = 0; ti < 4; ++ti) {
#pragma unroll
    for (int tj = 0; tj < 4; ++tj) {
      const int col = n0 + tj * 16 + cn;
      const float bv = bias[col];
#pragma unroll
      for (int r = 0; r < 8; ++r) {
        const size_t row = (size_t)m0 + ti * 16 + r + chi;
        float v = acc[ti][tj][r] + bv;
        if (EPI == EPI_BIAS_GELU_F16) {
          v = 0.5f * v * (1.0f + erff(v * 0.70710678118654752f)); // exact GELU
        }
        if (EPI == EPI_BIAS_RES_F32) {
          v += resid[row * N + col];
          ((float*)Cout)[row * N + col] = v;
        } else {
          ((_Float16*)Cout)[row * N + col] = (_Float16)v;
        }
      }
    }
  }
}

// ---------------------------------------------------------------------------
// Flash attention: one wave per (batch, head, 16-query tile).
// qkv layout: [M, 1344] f16, columns = Q(448) | K(448) | V(448),
// each section is HEADS x 64 (padded head dim, pad cols are exact zeros).
// Q was pre-scaled by 1/sqrt(54) at weight-pack time.
// o layout: [M, 448] f16.
// ---------------------------------------------------------------------------
__global__ __launch_bounds__(256) void attn_kernel(
    const _Float16* __restrict__ qkv, _Float16* __restrict__ o)
{
  __shared__ _Float16 pbuf[8][16 * 32];         // per-wave P re-swizzle tile
  const int wave = threadIdx.x >> 5;
  const int gw = blockIdx.x * 8 + wave;
  if (gw >= cfg::BATCH * cfg::HEADS * (cfg::SEQ / 16)) return;

  const int qt = gw & 31;
  const int rest = gw >> 5;
  const int h = rest % cfg::HEADS;
  const int b = rest / cfg::HEADS;
  const int q0 = qt * 16;
  const int ldq = cfg::NQKV;
  const size_t rowb = (size_t)b * cfg::SEQ;

  const _Float16* Q  = qkv + rowb * ldq + h * cfg::HSP;
  const _Float16* Kp = Q + cfg::NO;
  const _Float16* Vp = Q + 2 * cfg::NO;

  // Q fragments for this query tile (head dim 64 -> two K=32 fragments).
  const v16h qa0 = load_afrag(Q + (size_t)q0 * ldq,      ldq);
  const v16h qa1 = load_afrag(Q + (size_t)q0 * ldq + 32, ldq);

  const unsigned l = threadIdx.x & 31u;
  const int cn  = l & 15;
  const int chi = (l >> 4) << 3;                // row offset 0 or 8

  float rm[8], rl[8];
  v8f oacc[4];
#pragma unroll
  for (int r = 0; r < 8; ++r) { rm[r] = -1e30f; rl[r] = 0.0f; }
#pragma unroll
  for (int f = 0; f < 4; ++f)
#pragma unroll
    for (int r = 0; r < 8; ++r) oacc[f][r] = 0.0f;

  _Float16* myp = &pbuf[wave][0];

  for (int t0 = 0; t0 < q0 + 16; t0 += 32) {    // causal: keys up to q0+15
    // ---- scores: 16(q) x 32(t), via 4 WMMAs (K^T supplied as B fragments)
    v8f s0, s1;
#pragma unroll
    for (int r = 0; r < 8; ++r) { s0[r] = 0.0f; s1[r] = 0.0f; }
    {
      v16h kb;
      kb = load_bfrag(Kp + (size_t)t0 * ldq,             1, ldq);
      s0 = WMMA_F16(qa0, kb, s0);
      kb = load_bfrag(Kp + (size_t)t0 * ldq + 32,        1, ldq);
      s0 = WMMA_F16(qa1, kb, s0);
      kb = load_bfrag(Kp + (size_t)(t0 + 16) * ldq,      1, ldq);
      s1 = WMMA_F16(qa0, kb, s1);
      kb = load_bfrag(Kp + (size_t)(t0 + 16) * ldq + 32, 1, ldq);
      s1 = WMMA_F16(qa1, kb, s1);
    }

    // ---- causal mask + online softmax (row stats via intra-half shuffles)
#pragma unroll
    for (int r = 0; r < 8; ++r) {
      const int qrow = q0 + r + chi;
      float v0 = (t0 + cn      <= qrow) ? s0[r] : -1e30f;
      float v1 = (t0 + 16 + cn <= qrow) ? s1[r] : -1e30f;
      float mx = fmaxf(v0, v1);
#pragma unroll
      for (int d = 1; d < 16; d <<= 1) mx = fmaxf(mx, __shfl_xor(mx, d, 32));
      const float mnew  = fmaxf(rm[r], mx);
      const float alpha = __expf(rm[r] - mnew);
      const float p0 = __expf(v0 - mnew);
      const float p1 = __expf(v1 - mnew);
      float ps = p0 + p1;
#pragma unroll
      for (int d = 1; d < 16; d <<= 1) ps += __shfl_xor(ps, d, 32);
      rm[r] = mnew;
      rl[r] = rl[r] * alpha + ps;
#pragma unroll
      for (int f = 0; f < 4; ++f) oacc[f][r] *= alpha;
      // stash P (f16) for re-swizzle into A-fragment layout
      myp[(r + chi) * 32 + cn]      = (_Float16)p0;
      myp[(r + chi) * 32 + 16 + cn] = (_Float16)p1;
    }

    // wave-local LDS store->load ordering (DS ops are in-order per wave)
    asm volatile("s_wait_dscnt 0" ::: "memory");

    // ---- O += P (16x32) * V (32x64): 4 WMMAs
    const v16h pa = load_afrag(myp, 32);
#pragma unroll
    for (int f = 0; f < 4; ++f) {
      const v16h vb = load_bfrag(Vp + (size_t)t0 * ldq + f * 16, ldq, 1);
      oacc[f] = WMMA_F16(pa, vb, oacc[f]);
    }
  }

  // ---- normalize and write O tile ([M, 448], pad cols come out as zeros)
#pragma unroll
  for (int r = 0; r < 8; ++r) {
    const float inv = 1.0f / rl[r];
    const size_t row = rowb + q0 + r + chi;
#pragma unroll
    for (int f = 0; f < 4; ++f)
      o[row * cfg::NO + h * cfg::HSP + f * 16 + cn] = (_Float16)(oacc[f][r] * inv);
  }
}

// ---------------------------------------------------------------------------
// LayerNorm over 384 cols, fp32 in -> f16 out. One 128-thread block per row.
// ---------------------------------------------------------------------------
__global__ __launch_bounds__(128) void layernorm_f16_kernel(
    const float* __restrict__ x, const float* __restrict__ g,
    const float* __restrict__ bta, _Float16* __restrict__ out)
{
  __shared__ float red[128];
  const size_t row = blockIdx.x;
  const float* xr = x + row * cfg::EMBED;
  const int t = threadIdx.x;

  float v[3];
  float s = 0.0f;
#pragma unroll
  for (int i = 0; i < 3; ++i) { v[i] = xr[t + 128 * i]; s += v[i]; }
  red[t] = s; __syncthreads();
  for (int off = 64; off > 0; off >>= 1) {
    if (t < off) red[t] += red[t + off];
    __syncthreads();
  }
  const float mu = red[0] * (1.0f / cfg::EMBED);
  __syncthreads();

  float s2 = 0.0f;
#pragma unroll
  for (int i = 0; i < 3; ++i) { const float d = v[i] - mu; s2 += d * d; }
  red[t] = s2; __syncthreads();
  for (int off = 64; off > 0; off >>= 1) {
    if (t < off) red[t] += red[t + off];
    __syncthreads();
  }
  const float rs = rsqrtf(red[0] * (1.0f / cfg::EMBED) + 1e-5f);

#pragma unroll
  for (int i = 0; i < 3; ++i) {
    const int c = t + 128 * i;
    out[row * cfg::EMBED + c] = (_Float16)((v[i] - mu) * rs * g[c] + bta[c]);
  }
}

// ---------------------------------------------------------------------------
// Weight packing: fp32 -> f16, pad head size 54 -> 64, fold 1/sqrt(54) into Q,
// and swizzle into fragment-major [K/32][N][32] for b128 B-fragment loads.
// ---------------------------------------------------------------------------
__device__ __forceinline__ size_t bsw_index(int k, int n, int N) {
  return ((size_t)(k >> 5) * N + n) * 32 + (k & 31);
}

__global__ void pack_qkv_w_kernel(const float* __restrict__ wq,
                                  const float* __restrict__ wk,
                                  const float* __restrict__ wv,
                                  _Float16* __restrict__ Wqkv)
{
  const int idx = blockIdx.x * 256 + threadIdx.x;
  if (idx >= cfg::EMBED * cfg::NQKV) return;
  const int k   = idx / cfg::NQKV;              // embed (K) index
  const int col = idx % cfg::NQKV;
  const int sec = col / cfg::NO, rem = col % cfg::NO;
  const int hh = rem / cfg::HSP, dd = rem % cfg::HSP;
  float val = 0.0f;
  if (dd < cfg::HS) {
    const float* w = (sec == 0) ? wq : (sec == 1) ? wk : wv;
    val = w[((size_t)hh * cfg::EMBED + k) * cfg::HS + dd];
    if (sec == 0) val *= cfg::QSCALE;
  }
  Wqkv[bsw_index(k, col, cfg::NQKV)] = (_Float16)val;
}

__global__ void pack_qkv_b_kernel(const float* __restrict__ bq,
                                  const float* __restrict__ bk,
                                  const float* __restrict__ bv,
                                  float* __restrict__ bqkv)
{
  const int col = blockIdx.x * 256 + threadIdx.x;
  if (col >= cfg::NQKV) return;
  const int sec = col / cfg::NO, rem = col % cfg::NO;
  const int hh = rem / cfg::HSP, dd = rem % cfg::HSP;
  float val = 0.0f;
  if (dd < cfg::HS) {
    const float* bsrc = (sec == 0) ? bq : (sec == 1) ? bk : bv;
    val = bsrc[hh * cfg::HS + dd];
    if (sec == 0) val *= cfg::QSCALE;
  }
  bqkv[col] = val;
}

__global__ void pack_wo_kernel(const float* __restrict__ wo, _Float16* __restrict__ Wo)
{
  const int idx = blockIdx.x * 256 + threadIdx.x;
  if (idx >= cfg::NO * cfg::EMBED) return;
  const int k = idx / cfg::EMBED, c = idx % cfg::EMBED;  // k over padded 448
  const int hh = k / cfg::HSP, dd = k % cfg::HSP;
  const float val = (dd < cfg::HS)
      ? wo[((size_t)hh * cfg::HS + dd) * cfg::EMBED + c] : 0.0f;
  Wo[bsw_index(k, c, cfg::EMBED)] = (_Float16)val;
}

__global__ void pack_sw_kernel(const float* __restrict__ src,
                               _Float16* __restrict__ dst, int K, int N)
{
  const int i = blockIdx.x * 256 + threadIdx.x;
  if (i >= K * N) return;
  const int k = i / N, n = i % N;
  dst[bsw_index(k, n, N)] = (_Float16)src[i];
}

// ---------------------------------------------------------------------------
// Host-side launch
// ---------------------------------------------------------------------------
extern "C" void kernel_launch(void* const* d_in, const int* in_sizes, int n_in,
                              void* d_out, int out_size, void* d_ws, size_t ws_size,
                              hipStream_t stream) {
  using namespace cfg;
  const float* x     = (const float*)d_in[0];
  const float* wq    = (const float*)d_in[1];
  const float* bq    = (const float*)d_in[2];
  const float* wk    = (const float*)d_in[3];
  const float* bk    = (const float*)d_in[4];
  const float* wv    = (const float*)d_in[5];
  const float* bv    = (const float*)d_in[6];
  const float* wo    = (const float*)d_in[7];
  const float* bo    = (const float*)d_in[8];
  const float* w1    = (const float*)d_in[9];
  const float* b1    = (const float*)d_in[10];
  const float* w2    = (const float*)d_in[11];
  const float* b2    = (const float*)d_in[12];
  const float* ln1_g = (const float*)d_in[13];
  const float* ln1_b = (const float*)d_in[14];
  const float* ln2_g = (const float*)d_in[15];
  const float* ln2_b = (const float*)d_in[16];
  (void)in_sizes; (void)n_in; (void)out_size; (void)ws_size;

  // ---- workspace carve-out (256B aligned) ----
  char* ws = (char*)d_ws;
  size_t off = 0;
  auto carve = [&](size_t bytes) {
    void* p = ws + off;
    off = (off + bytes + 255) & ~(size_t)255;
    return p;
  };
  _Float16* A16   = (_Float16*)carve((size_t)M * EMBED * 2);   // h, reused as h2
  _Float16* QKVF  = (_Float16*)carve((size_t)M * FFN * 2);     // qkv, reused as ff1
  _Float16* Obuf  = (_Float16*)carve((size_t)M * NO * 2);
  float*    X1    = (float*)   carve((size_t)M * EMBED * 4);
  _Float16* Wqkv  = (_Float16*)carve((size_t)EMBED * NQKV * 2);
  float*    Bqkv  = (float*)   carve((size_t)NQKV * 4);
  _Float16* Wo    = (_Float16*)carve((size_t)NO * EMBED * 2);
  _Float16* W1h   = (_Float16*)carve((size_t)EMBED * FFN * 2);
  _Float16* W2h   = (_Float16*)carve((size_t)FFN * EMBED * 2);

  // ---- pack weights (fp32 -> f16 swizzled, pad 54 -> 64, fold q-scale) ----
  pack_qkv_w_kernel<<<(EMBED * NQKV + 255) / 256, 256, 0, stream>>>(wq, wk, wv, Wqkv);
  pack_qkv_b_kernel<<<(NQKV + 255) / 256, 256, 0, stream>>>(bq, bk, bv, Bqkv);
  pack_wo_kernel<<<(NO * EMBED + 255) / 256, 256, 0, stream>>>(wo, Wo);
  pack_sw_kernel<<<(EMBED * FFN + 255) / 256, 256, 0, stream>>>(w1, W1h, EMBED, FFN);
  pack_sw_kernel<<<(FFN * EMBED + 255) / 256, 256, 0, stream>>>(w2, W2h, FFN, EMBED);

  // ---- 1) LN1 -> h (f16) ----
  layernorm_f16_kernel<<<M, 128, 0, stream>>>(x, ln1_g, ln1_b, A16);

  // ---- 2) fused QKV projection: [M,384] x [384,1344] ----
  gemm_wmma_kernel<EPI_BIAS_F16, EMBED>
      <<<dim3((NQKV + 127) / 128, M / 256), 256, 0, stream>>>(
      A16, Wqkv, Bqkv, nullptr, QKVF, M, NQKV);

  // ---- 3) flash attention ----
  {
    const int waves = BATCH * HEADS * (SEQ / 16);     // 14336
    attn_kernel<<<(waves + 7) / 8, 256, 0, stream>>>(QKVF, Obuf);
  }

  // ---- 4) output projection + residual: x1 = x + O*Wo + bo ----
  gemm_wmma_kernel<EPI_BIAS_RES_F32, NO>
      <<<dim3((EMBED + 127) / 128, M / 256), 256, 0, stream>>>(
      Obuf, Wo, bo, x, X1, M, EMBED);

  // ---- 5) LN2 -> h2 (f16, reuse A16) ----
  layernorm_f16_kernel<<<M, 128, 0, stream>>>(X1, ln2_g, ln2_b, A16);

  // ---- 6) FFN1 + exact GELU: [M,384] x [384,2304] (ff1 reuses QKVF) ----
  gemm_wmma_kernel<EPI_BIAS_GELU_F16, EMBED>
      <<<dim3((FFN + 127) / 128, M / 256), 256, 0, stream>>>(
      A16, W1h, b1, nullptr, QKVF, M, FFN);

  // ---- 7) FFN2 + residual: out = x1 + ff1*W2 + b2 ----
  gemm_wmma_kernel<EPI_BIAS_RES_F32, FFN>
      <<<dim3((EMBED + 127) / 128, M / 256), 256, 0, stream>>>(
      QKVF, W2h, b2, X1, (float*)d_out, M, EMBED);
}
